// VectorQuantizerMultiHeadLegacy_89180700934751
// MI455X (gfx1250) — matrix operand order repaired and checked by reading, not compiled, not run on hardware
//
#include <hip/hip_runtime.h>
#include <hip/hip_bf16.h>

// CDNA5 / gfx1250 vector-quantizer (multi-head) using V_WMMA_F32_16X16X32_BF16
// with hi/lo bf16 split for ~fp32-accurate distances.

typedef __bf16 bf16_t;
typedef __attribute__((ext_vector_type(16))) __bf16 v16bf;
typedef __attribute__((ext_vector_type(8)))  float  v8f;

#define NUM_EMB 1024
#define HEADS   4
#define DHEAD   128
#define NROWS   32768
#define EDIM    512

// ---------------------------------------------------------------------------
// Prep: convert fp32 codebook -> bf16 hi/lo copies in workspace, and per-code
// squared norms (fp32, exact). 4096 code rows total.
// ---------------------------------------------------------------------------
__global__ void vq_prep(const float* __restrict__ emb,
                        bf16_t* __restrict__ ehi,
                        bf16_t* __restrict__ elo,
                        float*  __restrict__ wsq) {
  int rid = blockIdx.x * blockDim.x + threadIdx.x;   // 0 .. 4095
  if (rid >= HEADS * NUM_EMB) return;
  const float* src = emb + (size_t)rid * DHEAD;
  bf16_t* dh = ehi + (size_t)rid * DHEAD;
  bf16_t* dl = elo + (size_t)rid * DHEAD;
  float s = 0.f;
#pragma unroll 4
  for (int i = 0; i < DHEAD; i += 4) {
    float4 f = *(const float4*)(src + i);
    bf16_t h0 = (bf16_t)f.x, h1 = (bf16_t)f.y, h2 = (bf16_t)f.z, h3 = (bf16_t)f.w;
    dh[i + 0] = h0; dh[i + 1] = h1; dh[i + 2] = h2; dh[i + 3] = h3;
    dl[i + 0] = (bf16_t)(f.x - (float)h0);
    dl[i + 1] = (bf16_t)(f.y - (float)h1);
    dl[i + 2] = (bf16_t)(f.z - (float)h2);
    dl[i + 3] = (bf16_t)(f.w - (float)h3);
    s += f.x * f.x + f.y * f.y + f.z * f.z + f.w * f.w;
  }
  wsq[rid] = s;
}

// ---------------------------------------------------------------------------
// Main: one wave handles a 16-row tile of one head; sweeps all 1024 codes in
// 16-wide tiles via WMMA (3 bf16 WMMAs per K=32 chunk: hi*hi + hi*lo + lo*hi),
// tracks running argmin of x^2 + w^2 - 2 x.w in accumulator layout, then
// writes codes (int32), quantized rows (exact fp32 codebook values), and the
// commitment loss via one atomicAdd per wave.
// ---------------------------------------------------------------------------
__global__ __launch_bounds__(256)
void vq_main(const float*  __restrict__ x,
             const float*  __restrict__ emb,
             const bf16_t* __restrict__ ehi,
             const bf16_t* __restrict__ elo,
             const float*  __restrict__ wsq,
             float* __restrict__ out_loss,
             float* __restrict__ out_q,
             int*   __restrict__ out_codes) {
  const int lane = threadIdx.x & 31;
  const int wave = threadIdx.x >> 5;
  const int head = blockIdx.y;
  const int row0 = (blockIdx.x * 8 + wave) * 16;
  const int m    = lane & 15;   // row within tile (A) / col within tile (B,C)
  const int hh   = lane >> 4;   // lane half

  const size_t embRowBase = (size_t)head * NUM_EMB * DHEAD;
  const float* xrow = x + (size_t)(row0 + m) * EDIM + head * DHEAD;

  // ---- Load A tile (16x128) once; build bf16 hi/lo fragments. -------------
  // A-operand (16-bit, 16x32): lane half hh holds k in [16hh,16hh+16)
  // per chunk, packed as fragment elements e: k = kk + 16*hh + e.
  v16bf ahi[4], alo[4];
  float sq = 0.f;
#pragma unroll
  for (int c = 0; c < 4; ++c) {
    const int b1 = c * 32 + 16 * hh;
    float4 f0 = *(const float4*)(xrow + b1);
    float4 f1 = *(const float4*)(xrow + b1 + 4);
    float4 f2 = *(const float4*)(xrow + b1 + 8);
    float4 f3 = *(const float4*)(xrow + b1 + 12);
    float e[16] = {f0.x, f0.y, f0.z, f0.w, f1.x, f1.y, f1.z, f1.w,
                   f2.x, f2.y, f2.z, f2.w, f3.x, f3.y, f3.z, f3.w};
#pragma unroll
    for (int t = 0; t < 16; ++t) {
      float v = e[t];
      bf16_t h = (bf16_t)v;
      ahi[c][t] = h;
      alo[c][t] = (bf16_t)(v - (float)h);
      sq += v * v;
    }
  }
  // Full ||x_row||^2: lanes m and m+16 hold complementary k halves.
  sq += __shfl_xor(sq, 16, 32);
  float xsq[8];
#pragma unroll
  for (int r = 0; r < 8; ++r) xsq[r] = __shfl(sq, r + 8 * hh, 32);

  float minv[8];
  int   mini[8];
#pragma unroll
  for (int r = 0; r < 8; ++r) { minv[r] = 3.402823466e38f; mini[r] = 0; }

  // ---- Sweep 64 code tiles of 16. -----------------------------------------
  for (int t = 0; t < 64; ++t) {
    const int col0 = t * 16;
    // B-operand (32x16 bf16): lane = n + 16*hh holds k = kk + 16*hh + e,
    // i.e. 16 contiguous bf16 of codebook row (col0+n) -> direct v16bf load.
    const bf16_t* bh = ehi + embRowBase + (size_t)(col0 + m) * DHEAD + 16 * hh;
    const bf16_t* bl = elo + embRowBase + (size_t)(col0 + m) * DHEAD + 16 * hh;
    v8f acc = {0.f, 0.f, 0.f, 0.f, 0.f, 0.f, 0.f, 0.f};
#pragma unroll
    for (int c = 0; c < 4; ++c) {
      v16bf Bh = *(const v16bf*)(bh + c * 32);
      v16bf Bl = *(const v16bf*)(bl + c * 32);
      acc = __builtin_amdgcn_wmma_f32_16x16x32_bf16(false, ahi[c], false, Bh,
                                                    (short)0, acc, false, false);
      acc = __builtin_amdgcn_wmma_f32_16x16x32_bf16(false, ahi[c], false, Bl,
                                                    (short)0, acc, false, false);
      acc = __builtin_amdgcn_wmma_f32_16x16x32_bf16(false, alo[c], false, Bh,
                                                    (short)0, acc, false, false);
    }
    const float wn  = wsq[head * NUM_EMB + col0 + m];
    const int   idx = col0 + m;
#pragma unroll
    for (int r = 0; r < 8; ++r) {
      // C layout: VGPR r, lane -> (M = r + 8*hh, N = lane&15)
      float d = xsq[r] + wn - 2.0f * acc[r];
      if (d < minv[r]) { minv[r] = d; mini[r] = idx; }
    }
  }

  // ---- Argmin across the 16 lanes of each half (low-index tie-break). -----
#pragma unroll
  for (int r = 0; r < 8; ++r) {
    float v = minv[r];
    int   i = mini[r];
#pragma unroll
    for (int off = 1; off < 16; off <<= 1) {
      float vo = __shfl_xor(v, off, 32);
      int   io = __shfl_xor(i, off, 32);
      if (vo < v || (vo == v && io < i)) { v = vo; i = io; }
    }
    mini[r] = i;
  }

  // codes[n][h]: lane 0 covers rows row0..row0+7, lane 16 covers +8..+15
  if (m == 0) {
#pragma unroll
    for (int r = 0; r < 8; ++r)
      out_codes[(size_t)(row0 + 8 * hh + r) * HEADS + head] = mini[r];
  }

  // ---- Quantized copy (exact fp32 codebook rows) + commitment loss. -------
  float lsum = 0.f;
#pragma unroll
  for (int mm = 0; mm < 16; ++mm) {
    const int code = __shfl(mini[mm & 7], (mm >> 3) * 16, 32);
    const float* qrow = emb + embRowBase + (size_t)code * DHEAD + lane * 4;
    const float* xr   = x + (size_t)(row0 + mm) * EDIM + head * DHEAD + lane * 4;
    float* orow = out_q + (size_t)(row0 + mm) * EDIM + head * DHEAD + lane * 4;
    float4 q  = *(const float4*)qrow;
    float4 xv = *(const float4*)xr;
    // out_q region starts at d_out+1 float -> only 4B aligned: scalar stores.
    orow[0] = q.x; orow[1] = q.y; orow[2] = q.z; orow[3] = q.w;
    float d0 = q.x - xv.x, d1 = q.y - xv.y, d2 = q.z - xv.z, d3 = q.w - xv.w;
    lsum += d0 * d0 + d1 * d1 + d2 * d2 + d3 * d3;
  }
#pragma unroll
  for (int off = 1; off < 32; off <<= 1) lsum += __shfl_xor(lsum, off, 32);
  if (lane == 0)
    atomicAdd(out_loss, lsum * (0.25f / ((float)NROWS * (float)EDIM)));
}

// ---------------------------------------------------------------------------
extern "C" void kernel_launch(void* const* d_in, const int* in_sizes, int n_in,
                              void* d_out, int out_size, void* d_ws, size_t ws_size,
                              hipStream_t stream) {
  const float* x   = (const float*)d_in[0];   // (32768, 512) fp32
  const float* emb = (const float*)d_in[1];   // (4, 1024, 128) fp32
  float* out = (float*)d_out;

  // d_out layout (return order): [0] loss, [1 .. 1+N*512) quantized_st,
  // then codes (N,4) as int32 bit-pattern in the tail.
  float* out_q     = out + 1;
  int*   out_codes = (int*)(out + 1 + (size_t)NROWS * EDIM);

  // Workspace: bf16 hi codebook (1MB) | bf16 lo codebook (1MB) | w_sq (16KB)
  bf16_t* ehi = (bf16_t*)d_ws;
  bf16_t* elo = ehi + (size_t)HEADS * NUM_EMB * DHEAD;
  float*  wsq = (float*)(elo + (size_t)HEADS * NUM_EMB * DHEAD);

  hipMemsetAsync(d_out, 0, sizeof(float), stream);  // zero the loss accumulator

  vq_prep<<<dim3((HEADS * NUM_EMB) / 256), 256, 0, stream>>>(emb, ehi, elo, wsq);

  // 2048 row tiles of 16, 8 waves (256 threads) per block, one grid.y per head
  vq_main<<<dim3(NROWS / (16 * 8), HEADS), 256, 0, stream>>>(
      x, emb, ehi, elo, wsq, out, out_q, out_codes);
}
